// RelationPrediction_4045859192984
// MI455X (gfx1250) — matrix operand order repaired and checked by reading, not compiled, not run on hardware
//
#include <hip/hip_runtime.h>

// Problem dims (from reference)
#define BATCH 32
#define CDIM  256
#define TDIM  256
#define VDIM  34
#define DDIM  64
#define ROWS_PER_BATCH (CDIM * TDIM)            // 65536 rows of 34 floats per batch
#define CHUNK_ROWS 128
#define CHUNKS_PER_BLOCK 8
#define BLOCKS_PER_BATCH (ROWS_PER_BATCH / (CHUNK_ROWS * CHUNKS_PER_BLOCK)) // 64
#define LDS_STRIDE 72                            // halves per LDS row (16B aligned, low-conflict)
#define PAIRS_PER_CHUNK (CHUNK_ROWS * VDIM / 2)  // 2176 = 8*256 + 128

typedef __attribute__((ext_vector_type(16))) __bf16 v16bf;
typedef __attribute__((ext_vector_type(8)))  __bf16 v8bf;
typedef __attribute__((ext_vector_type(8)))  float  v8f;

__device__ __forceinline__ unsigned pk_bf16x2(float a, float b) {
    unsigned ua = __builtin_bit_cast(unsigned, a);
    unsigned ub = __builtin_bit_cast(unsigned, b);
    ua = (ua + 0x7FFFu + ((ua >> 16) & 1u)) >> 16;      // RNE
    ub =  ub + 0x7FFFu + ((ub >> 16) & 1u);
    return (ub & 0xFFFF0000u) | ua;
}
__device__ __forceinline__ unsigned short f2bf(float a) {
    unsigned ua = __builtin_bit_cast(unsigned, a);
    return (unsigned short)((ua + 0x7FFFu + ((ua >> 16) & 1u)) >> 16);
}

// ---------------------------------------------------------------------------
// Kernel 1: temporal mean pool.  x[b][c][t][v] -> pool[b][c][v]
// One block per (b,c): 256*34 = 8704 contiguous floats, fully coalesced loads.
// flat idx = tid + 256*j ; v = idx % 34 ; pairs j and j+17 share the same v.
// ---------------------------------------------------------------------------
__global__ void __launch_bounds__(256)
pool_kernel(const float* __restrict__ x, float* __restrict__ pool)
{
    const int tid = threadIdx.x;
    const float* base = x + (size_t)blockIdx.x * (TDIM * VDIM);
    __shared__ float acc[VDIM];
    if (tid < VDIM) acc[tid] = 0.0f;
    __syncthreads();

    int v = tid % VDIM;
#pragma unroll
    for (int j = 0; j < 17; ++j) {
        float s = base[tid + 256 * j] + base[tid + 256 * (j + 17)];
        atomicAdd(&acc[v], s);                 // ds_add_f32; lanes hit distinct banks
        v += 18; if (v >= VDIM) v -= VDIM;     // (v + 256) mod 34
    }
    __syncthreads();
    if (tid < VDIM)
        pool[(size_t)blockIdx.x * VDIM + tid] = acc[tid] * (1.0f / (float)TDIM);
}

// ---------------------------------------------------------------------------
// Kernel 2: per-batch q/k projection, scores, softmax, +bias, then write
// C_adaptive pre-swizzled into the WMMA bf16 B-operand layout:
//   fragment f = s*3 + nt  (k-step s in {0,1}, n-tile nt in {0,1,2})
//   lane L, element e  ->  K = 32*s + 8*(L/16) + (e%8) + 16*(e/8) ,  N = 16*nt + (L%16)
//   zero-padded for K>=34 or N>=34.
// ---------------------------------------------------------------------------
__global__ void __launch_bounds__(256)
attn_kernel(const float* __restrict__ pool, const float* __restrict__ Bbias,
            const float* __restrict__ Wq, const float* __restrict__ bq,
            const float* __restrict__ Wk, const float* __restrict__ bk,
            unsigned short* __restrict__ bswz)
{
    const int b   = blockIdx.x;
    const int tid = threadIdx.x;
    __shared__ float sp[CDIM * VDIM];   // pooled features [c][v]
    __shared__ float sq[DDIM * VDIM];   // q [d][v]
    __shared__ float sk[DDIM * VDIM];   // k [d][v]
    __shared__ float sc[VDIM * VDIM];   // C_adaptive [v][w]

    for (int i = tid; i < CDIM * VDIM; i += 256)
        sp[i] = pool[(size_t)b * (CDIM * VDIM) + i];
    __syncthreads();

    // q = Wq @ pool + bq ;  k = Wk @ pool + bk     (2176 outputs each)
    for (int i = tid; i < DDIM * VDIM; i += 256) {
        const int d = i / VDIM, v = i - d * VDIM;
        float aq = bq[d], ak = bk[d];
        const float* wqr = Wq + (size_t)d * CDIM;
        const float* wkr = Wk + (size_t)d * CDIM;
        for (int c = 0; c < CDIM; ++c) {
            const float p = sp[c * VDIM + v];
            aq = fmaf(wqr[c], p, aq);
            ak = fmaf(wkr[c], p, ak);
        }
        sq[i] = aq; sk[i] = ak;
    }
    __syncthreads();

    // scores -> softmax(row) -> + bias
    if (tid < VDIM) {
        float row[VDIM];
#pragma unroll 1
        for (int w = 0; w < VDIM; ++w) {
            float s = 0.0f;
            for (int d = 0; d < DDIM; ++d)
                s = fmaf(sq[d * VDIM + tid], sk[d * VDIM + w], s);
            row[w] = s * 0.125f;                        // 1/sqrt(64)
        }
        float mx = row[0];
#pragma unroll
        for (int w = 1; w < VDIM; ++w) mx = fmaxf(mx, row[w]);
        float sum = 0.0f;
#pragma unroll
        for (int w = 0; w < VDIM; ++w) { row[w] = __expf(row[w] - mx); sum += row[w]; }
        const float inv = 1.0f / sum;
#pragma unroll
        for (int w = 0; w < VDIM; ++w)
            sc[tid * VDIM + w] = row[w] * inv + Bbias[tid * VDIM + w];
    }
    __syncthreads();

    // swizzle into B-operand fragments: 6 frags * 32 lanes * 16 elems = 3072 bf16
    for (int i = tid; i < 6 * 32 * 16; i += 256) {
        const int e  = i & 15;
        const int L  = (i >> 4) & 31;
        const int f  = i >> 9;
        const int s  = f / 3, nt = f - s * 3;
        const int K  = 32 * s + 8 * (L >> 4) + (e & 7) + 16 * (e >> 3);
        const int N  = 16 * nt + (L & 15);
        const float val = (K < VDIM && N < VDIM) ? sc[K * VDIM + N] : 0.0f;
        bswz[(((size_t)b * 6 + f) * 32 + L) * 16 + e] = f2bf(val);
    }
}

// ---------------------------------------------------------------------------
// Kernel 3: out[b, row, :] = x[b, row, :] @ C_adaptive[b]   (WMMA bf16)
// 8 wave32 waves per block; each wave owns a 16-row tile of a 128-row chunk.
// A fragments built from LDS per the 16-bit A layout:
//   lane<16 : row = lane,    K in {c..c+7, c+16..c+23}, c = 32*s
//   lane>=16: row = lane-16, same with c = 32*s + 8
// ---------------------------------------------------------------------------
__global__ void __launch_bounds__(256)
agg_wmma_kernel(const float* __restrict__ x,
                const unsigned short* __restrict__ bswz,
                float* __restrict__ out)
{
    __shared__ __align__(16) unsigned short tile[CHUNK_ROWS * LDS_STRIDE];
    const int tid  = threadIdx.x;
    const int lane = tid & 31;                 // wave32
    const int wv   = tid >> 5;
    const int b      = blockIdx.x / BLOCKS_PER_BATCH;
    const int blkInB = blockIdx.x % BLOCKS_PER_BATCH;

    // B fragments for this batch (identical per wave, per-lane 32B coalesced)
    v16bf bf[6];
#pragma unroll
    for (int f = 0; f < 6; ++f)
        bf[f] = *(const v16bf*)(bswz + (((size_t)b * 6 + f) * 32 + lane) * 16);

    // zero LDS once: pad columns (>=34) and pad K range stay zero forever
    unsigned* tz = (unsigned*)tile;
    for (int i = tid; i < CHUNK_ROWS * LDS_STRIDE / 2; i += 256) tz[i] = 0u;
    __syncthreads();

    const int m  = lane & 15;
    const int hi = lane >> 4;

    for (int ch = 0; ch < CHUNKS_PER_BLOCK; ++ch) {
        const size_t rowBase = (size_t)b * ROWS_PER_BATCH
                             + (size_t)blkInB * (CHUNK_ROWS * CHUNKS_PER_BLOCK)
                             + (size_t)ch * CHUNK_ROWS;
        const float* src = x + rowBase * VDIM;

        // prefetch next chunk (global_prefetch_b8)
        if (ch + 1 < CHUNKS_PER_BLOCK) {
            const char* nx = (const char*)(src + CHUNK_ROWS * VDIM);
            if (tid * 64 < CHUNK_ROWS * VDIM * 4)
                __builtin_prefetch(nx + (size_t)tid * 64, 0, 3);
        }

        // stage: coalesced b64 loads of f32 pairs -> packed bf16 -> LDS.
        // 34 is even, so a pair never crosses a row; 17 pairs per row.
        // 2176 pairs = 8 full uniform rounds of 256 + a 128-thread tail, so the
        // 8 main loads are branch-free (single s_clause of global_load_b64).
        const float2* s2 = (const float2*)src;
        float2 pv[8];
#pragma unroll
        for (int j = 0; j < 8; ++j)
            pv[j] = s2[tid + 256 * j];
        float2 tv;
        if (tid < PAIRS_PER_CHUNK - 8 * 256)               // tail: 128 pairs
            tv = s2[tid + 8 * 256];
#pragma unroll
        for (int j = 0; j < 8; ++j) {
            const int p = tid + 256 * j;
            const int r = p / 17;
            const int c = (p - r * 17) * 2;
            ((unsigned*)tile)[(r * LDS_STRIDE + c) >> 1] = pk_bf16x2(pv[j].x, pv[j].y);
        }
        if (tid < PAIRS_PER_CHUNK - 8 * 256) {
            const int p = tid + 8 * 256;
            const int r = p / 17;
            const int c = (p - r * 17) * 2;
            ((unsigned*)tile)[(r * LDS_STRIDE + c) >> 1] = pk_bf16x2(tv.x, tv.y);
        }
        __syncthreads();

        // A fragments for this wave's 16-row tile (4x ds_load_b128 per lane)
        const unsigned short* rowp = tile + (wv * 16 + m) * LDS_STRIDE + hi * 8;
        const v8bf a0lo = *(const v8bf*)(rowp);
        const v8bf a0hi = *(const v8bf*)(rowp + 16);
        const v8bf a1lo = *(const v8bf*)(rowp + 32);
        const v8bf a1hi = *(const v8bf*)(rowp + 48);
        const v16bf A0 = __builtin_shufflevector(a0lo, a0hi,
                           0,1,2,3,4,5,6,7,8,9,10,11,12,13,14,15);
        const v16bf A1 = __builtin_shufflevector(a1lo, a1hi,
                           0,1,2,3,4,5,6,7,8,9,10,11,12,13,14,15);

        v8f c0 = {}, c1 = {}, c2 = {};
        c0 = __builtin_amdgcn_wmma_f32_16x16x32_bf16(false, A0, false, bf[0], (short)0, c0, false, false);
        c1 = __builtin_amdgcn_wmma_f32_16x16x32_bf16(false, A0, false, bf[1], (short)0, c1, false, false);
        c2 = __builtin_amdgcn_wmma_f32_16x16x32_bf16(false, A0, false, bf[2], (short)0, c2, false, false);
        c0 = __builtin_amdgcn_wmma_f32_16x16x32_bf16(false, A1, false, bf[3], (short)0, c0, false, false);
        c1 = __builtin_amdgcn_wmma_f32_16x16x32_bf16(false, A1, false, bf[4], (short)0, c1, false, false);
        c2 = __builtin_amdgcn_wmma_f32_16x16x32_bf16(false, A1, false, bf[5], (short)0, c2, false, false);

        // store: C/D layout -> VGPR r holds M = r + 8*hi, N = (lane%16) + 16*nt
        float* orow = out + (rowBase + (size_t)wv * 16 + (size_t)hi * 8) * VDIM;
#pragma unroll
        for (int r = 0; r < 8; ++r) {
            float* o = orow + (size_t)r * VDIM;
            o[m]      = c0[r];
            o[16 + m] = c1[r];
        }
        if (m < 2) {
#pragma unroll
            for (int r = 0; r < 8; ++r)
                orow[(size_t)r * VDIM + 32 + m] = c2[r];
        }
        __syncthreads();
    }
}

// ---------------------------------------------------------------------------
extern "C" void kernel_launch(void* const* d_in, const int* in_sizes, int n_in,
                              void* d_out, int out_size, void* d_ws, size_t ws_size,
                              hipStream_t stream) {
    (void)in_sizes; (void)n_in; (void)out_size; (void)ws_size;
    const float* x     = (const float*)d_in[0];
    const float* Bbias = (const float*)d_in[1];
    const float* Wq    = (const float*)d_in[2];
    const float* bq    = (const float*)d_in[3];
    const float* Wk    = (const float*)d_in[4];
    const float* bk    = (const float*)d_in[5];
    float* out = (float*)d_out;

    // workspace layout: pooled features, then swizzled bf16 B fragments
    float* d_pool = (float*)d_ws;                                    // 32*256*34 f32 = 1,114,112 B
    unsigned short* d_bswz =
        (unsigned short*)((char*)d_ws + (size_t)BATCH * CDIM * VDIM * sizeof(float)); // 196,608 B

    pool_kernel<<<BATCH * CDIM, 256, 0, stream>>>(x, d_pool);
    attn_kernel<<<BATCH, 256, 0, stream>>>(d_pool, Bbias, Wq, bq, Wk, bk, d_bswz);
    agg_wmma_kernel<<<BATCH * BLOCKS_PER_BATCH, 256, 0, stream>>>(x, d_bswz, out);
}
